// TwoAttnGINConvNet_21492016349784
// MI455X (gfx1250) — matrix operand order repaired and checked by reading, not compiled, not run on hardware
//
#include <hip/hip_runtime.h>
#include <hip/hip_bf16.h>

typedef __bf16 bf16;
typedef __attribute__((ext_vector_type(16))) __bf16 bf16x16;
typedef __attribute__((ext_vector_type(8)))  float  f32x8;

namespace {
constexpr int kN     = 10240;   // nodes
constexpr int kB     = 256;     // graphs / batch
constexpr int kNper  = 40;      // nodes per graph (batch is sorted, exactly 40 each)
constexpr int kE     = 40960;   // edges
constexpr int kNFXD  = 78;
constexpr int kDim   = 121;
constexpr int kDimP  = 128;     // padded feature stride
constexpr int kSeq   = 1000;
constexpr int kNfilt = 32;
constexpr int kKw    = 8;       // conv kernel width
constexpr int kL2    = 114;
constexpr int kVocab = 26;
constexpr int kKc1   = 224;     // conv1 folded K: 26*8=208 (+bias col 208), padded to 224
}

union FragAB { bf16x16 v; uint4 q[2]; };

// ---------------------------------------------------------------------------
// Generic bf16 WMMA GEMM: out[M,Nvalid] = A[M,Kp] @ W[K,N] (+bias)(relu)(bn)
// A: bf16 row-major, K padded to Kp (mult of 32), zeros in pad.
// Bt: bf16 [Np, Kp] = W^T padded with zeros (Np mult of 64).
// One wave computes a 16x64 strip: 4 accumulators share one A fragment per
// K-step (4x fewer A reads, 4 v_wmma per A load).
// Fragment layouts per CDNA5 ISA 7.12.2 (wave32):
//   A 16x32: lane l: row=l&15; elems 0..7 -> K = k0+8*(l>>4)+e ; elems 8..15 -> +16
//   B 32x16 (from transposed Bt): lane l: col=l&15; elems 0..7 -> K = k0+8*(l>>4)+e ; elems 8..15 -> +16
//   C/D: lane l: col = l&15 ; vgpr j: row = j + 8*(l>>4)
// ---------------------------------------------------------------------------
__global__ void k_gemm(const bf16* __restrict__ A, const bf16* __restrict__ Bt,
                       int M, int Kp, int Np, int Nvalid,
                       const float* __restrict__ bias, int relu,
                       const float* __restrict__ bng, const float* __restrict__ bnb,
                       const float* __restrict__ bnm, const float* __restrict__ bnv,
                       float* __restrict__ outF, int ldF,
                       bf16* __restrict__ outB, int ldB)
{
    const int nstrips = Np >> 6;            // 4 consecutive 16-wide n-tiles per wave
    const int mtiles  = M >> 4;
    int w = blockIdx.x * (blockDim.x >> 5) + (threadIdx.x >> 5);
    if (w >= mtiles * nstrips) return;
    const int mt = w / nstrips, ns = w % nstrips;
    const int lane = threadIdx.x & 31;
    const int l16 = lane & 15, half = lane >> 4;

    const bf16* arow  = A  + (size_t)(mt * 16 + l16) * Kp;
    const bf16* brow0 = Bt + (size_t)(ns * 64 + l16) * Kp;

    f32x8 c[4];
#pragma unroll
    for (int t = 0; t < 4; ++t) c[t] = (f32x8){0.f,0.f,0.f,0.f,0.f,0.f,0.f,0.f};

    const int nk = Kp >> 5;
    for (int kk = 0; kk < nk; ++kk) {
        const int k0 = kk << 5;
        FragAB fa;
        fa.q[0] = *reinterpret_cast<const uint4*>(arow + k0 + 8 * half);
        fa.q[1] = *reinterpret_cast<const uint4*>(arow + k0 + 16 + 8 * half);
        if (kk + 1 < nk) {                   // CDNA5 global_prefetch_b8 path
            __builtin_prefetch(arow + k0 + 32, 0, 1);
            __builtin_prefetch(brow0 + k0 + 32, 0, 1);
        }
#pragma unroll
        for (int t = 0; t < 4; ++t) {
            const bf16* br = brow0 + (size_t)t * 16 * Kp + k0;
            FragAB fb;
            fb.q[0] = *reinterpret_cast<const uint4*>(br + 8 * half);
            fb.q[1] = *reinterpret_cast<const uint4*>(br + 16 + 8 * half);
            c[t] = __builtin_amdgcn_wmma_f32_16x16x32_bf16(false, fa.v, false, fb.v,
                                                           (short)0, c[t], false, false);
        }
    }

#pragma unroll
    for (int t = 0; t < 4; ++t) {
        const int n = ns * 64 + t * 16 + l16;
        const float bv = (bias && n < Nvalid) ? bias[n] : 0.f;
        float g = 1.f, b0 = 0.f, mm = 0.f, inv = 1.f;
        if (bng && n < Nvalid) {
            g = bng[n]; b0 = bnb[n]; mm = bnm[n];
            inv = rsqrtf(bnv[n] + 1e-5f);
        }
#pragma unroll
        for (int j = 0; j < 8; ++j) {
            const int r = mt * 16 + j + 8 * half;
            float v = c[t][j] + bv;
            if (relu) v = v > 0.f ? v : 0.f;
            if (bng)  v = (v - mm) * inv * g + b0;
            if (outF && n < Nvalid && r < M) outF[(size_t)r * ldF + n] = v;
            if (outB && r < M) outB[(size_t)r * ldB + n] = (n < Nvalid) ? (bf16)v : (bf16)0.f;
        }
    }
}

// --- weight convert: Wt[Np,Kp] (bf16, zero padded) = transpose(W[K,N]) -------
__global__ void k_convW(bf16* __restrict__ Wt, const float* __restrict__ W,
                        int K, int N, int Kp, int Np)
{
    int i = blockIdx.x * blockDim.x + threadIdx.x;
    if (i >= Np * Kp) return;
    int n = i / Kp, k = i % Kp;
    Wt[i] = (n < N && k < K) ? (bf16)W[(size_t)k * N + n] : (bf16)0.f;
}

// --- activation convert: Ab[M,Kp] (bf16, zero padded) = src[M,*] f32 ---------
__global__ void k_convA(bf16* __restrict__ dst, const float* __restrict__ src,
                        int srcStride, int M, int K, int Kp)
{
    int i = blockIdx.x * blockDim.x + threadIdx.x;
    if (i >= M * Kp) return;
    int m = i / Kp, k = i % Kp;
    dst[i] = (k < K) ? (bf16)src[(size_t)m * srcStride + k] : (bf16)0.f;
}

// --- agg = h (copy, stride128 dest) -----------------------------------------
__global__ void k_copy(float* __restrict__ agg, const float* __restrict__ src,
                       int srcStride, int C)
{
    int b = blockIdx.x, c = threadIdx.x;
    if (c < C) agg[(size_t)b * kDimP + c] = src[(size_t)b * srcStride + c];
}

// --- agg[dst[e]] += h[src[e]]  (GIN neighbor sum) ---------------------------
__global__ void k_scatter(float* __restrict__ agg, const float* __restrict__ h,
                          int srcStride, const int* __restrict__ edge, int C)
{
    int e = blockIdx.x, c = threadIdx.x;
    if (c >= C) return;
    int s = edge[e], d = edge[kE + e];
    atomicAdd(&agg[(size_t)d * kDimP + c], h[(size_t)s * srcStride + c]);
}

// --- conv1 folded operand: Abig[(b,f), v*8+k] = sum_{s: tgt[b,s]=v} w1[f,s,k]
//     col 208 carries bias (TabT col 208 == 1).  ------------------------------
__global__ void k_build_A1(bf16* __restrict__ Abig, const float* __restrict__ w1,
                           const float* __restrict__ b1, const int* __restrict__ target)
{
    __shared__ int   tg[kSeq];
    __shared__ float acc[kVocab * 256];
    const int b = blockIdx.x, tid = threadIdx.x;
    const int f = tid >> 3, k = tid & 7;
    for (int i = tid; i < kSeq; i += 256) tg[i] = target[b * kSeq + i];
    for (int v = 0; v < kVocab; ++v) acc[v * 256 + tid] = 0.f;
    __syncthreads();
    const float* wrow = w1 + (size_t)f * kSeq * kKw + k;
    for (int s = 0; s < kSeq; ++s)
        acc[tg[s] * 256 + tid] += wrow[s * kKw];
    const size_t row = (size_t)(b * kNfilt + f) * kKc1;
    for (int v = 0; v < kVocab; ++v)
        Abig[row + v * 8 + k] = (bf16)acc[v * 256 + tid];
    if (k == 0) Abig[row + 208] = (bf16)b1[f];
    for (int c = 209 + k; c < kKc1; c += 8) Abig[row + c] = (bf16)0.f;
}

// --- TabT[l, v*8+k] = table[v, l+k]; col 208 = 1 (bias); zero padded --------
__global__ void k_build_TabT(bf16* __restrict__ TabT, const float* __restrict__ emb)
{
    int l = blockIdx.x, c = threadIdx.x;   // grid 128 x 224
    bf16 val = (bf16)0.f;
    if (l < kDim) {
        if (c < kVocab * 8) { int v = c >> 3, k = c & 7; val = (bf16)emb[v * 128 + l + k]; }
        else if (c == 208) val = (bf16)1.0f;
    }
    TabT[l * kKc1 + c] = val;
}

// --- attention 1: q=c1f[b] (32x121), kv=h3 rows (40x121) --------------------
__global__ void k_attn1(float* __restrict__ o1f, const float* __restrict__ c1f,
                        const float* __restrict__ h3)
{
    __shared__ float kv[kNper * kDimP];
    __shared__ float qs[kNfilt * kDimP];
    __shared__ float sc[kNfilt * kNper];
    const int b = blockIdx.x, tid = threadIdx.x;
    for (int i = tid; i < kNper * kDimP; i += 256) {
        int p = i >> 7, c = i & 127;
        kv[i] = (c < kDim) ? h3[(size_t)(b * kNper + p) * kDimP + c] : 0.f;
    }
    for (int i = tid; i < kNfilt * kDimP; i += 256) {
        int q = i >> 7, c = i & 127;
        qs[i] = (c < kDim) ? c1f[(size_t)(b * kNfilt + q) * kDim + c] : 0.f;
    }
    __syncthreads();
    for (int p = tid; p < kNfilt * kNper; p += 256) {
        int qi = p / kNper, ki = p % kNper;
        float s = 0.f;
        for (int c = 0; c < kDim; ++c) s += qs[qi * kDimP + c] * kv[ki * kDimP + c];
        sc[p] = s * (1.f / 11.f);              // 1/sqrt(121)
    }
    __syncthreads();
    if (tid < kNfilt) {
        float* row = sc + tid * kNper;
        float m = row[0];
        for (int i = 1; i < kNper; ++i) m = fmaxf(m, row[i]);
        float sum = 0.f;
        for (int i = 0; i < kNper; ++i) { float e = __expf(row[i] - m); row[i] = e; sum += e; }
        float r = 1.f / sum;
        for (int i = 0; i < kNper; ++i) row[i] *= r;
    }
    __syncthreads();
    for (int i = tid; i < kNfilt * kDim; i += 256) {
        int qi = i / kDim, l = i % kDim;
        float a = 0.f;
        for (int ki = 0; ki < kNper; ++ki) a += sc[qi * kNper + ki] * kv[ki * kDimP + l];
        o1f[(size_t)(b * kNfilt + qi) * kDim + l] = a;
    }
}

// --- conv2 (VALU, 0.48 GF) fused with attention 2; emits o2 in bf16 ---------
__global__ void k_conv2_attn2(bf16* __restrict__ o2bf, const float* __restrict__ o1f,
                              const float* __restrict__ h5, const float* __restrict__ w2,
                              const float* __restrict__ b2)
{
    __shared__ float o1s[kNfilt * kDimP];
    __shared__ float c2s[kNfilt * kL2];
    __shared__ float kv[kNper * kL2];
    __shared__ float sc[kNfilt * kNper];
    const int b = blockIdx.x, tid = threadIdx.x;
    for (int i = tid; i < kNfilt * kDimP; i += 256) {
        int q = i >> 7, c = i & 127;
        o1s[i] = (c < kDim) ? o1f[(size_t)(b * kNfilt + q) * kDim + c] : 0.f;
    }
    for (int i = tid; i < kNper * kL2; i += 256) {
        int p = i / kL2, l = i % kL2;
        kv[i] = h5[(size_t)(b * kNper + p) * kDimP + l];     // hd2 = h5[..., :114]
    }
    __syncthreads();
    for (int p = tid; p < kNfilt * kL2; p += 256) {
        int f = p / kL2, l = p % kL2;
        float acc = b2[f];
        for (int c = 0; c < kNfilt; ++c) {
            const float* w = w2 + (size_t)(f * kNfilt + c) * kKw;
            const float* o = o1s + c * kDimP + l;
#pragma unroll
            for (int k = 0; k < kKw; ++k) acc += o[k] * w[k];
        }
        c2s[p] = acc;
    }
    __syncthreads();
    const float scale = rsqrtf((float)kL2);
    for (int p = tid; p < kNfilt * kNper; p += 256) {
        int qi = p / kNper, ki = p % kNper;
        float s = 0.f;
        for (int c = 0; c < kL2; ++c) s += c2s[qi * kL2 + c] * kv[ki * kL2 + c];
        sc[p] = s * scale;
    }
    __syncthreads();
    if (tid < kNfilt) {
        float* row = sc + tid * kNper;
        float m = row[0];
        for (int i = 1; i < kNper; ++i) m = fmaxf(m, row[i]);
        float sum = 0.f;
        for (int i = 0; i < kNper; ++i) { float e = __expf(row[i] - m); row[i] = e; sum += e; }
        float r = 1.f / sum;
        for (int i = 0; i < kNper; ++i) row[i] *= r;
    }
    __syncthreads();
    for (int i = tid; i < kNfilt * kL2; i += 256) {
        int qi = i / kL2, l = i % kL2;
        float a = 0.f;
        for (int ki = 0; ki < kNper; ++ki) a += sc[qi * kNper + ki] * kv[ki * kL2 + l];
        o2bf[(size_t)b * (kNfilt * kL2) + qi * kL2 + l] = (bf16)a;   // reshape order f*114+l
    }
}

// --- global add pool over the 40 nodes of each graph -> bf16 padded ---------
__global__ void k_pool(bf16* __restrict__ xg, const float* __restrict__ h5)
{
    int b = blockIdx.x, c = threadIdx.x;   // 256 x 128
    float s = 0.f;
    if (c < kDim)
        for (int p = 0; p < kNper; ++p) s += h5[(size_t)(b * kNper + p) * kDimP + c];
    xg[b * kDimP + c] = (c < kDim) ? (bf16)s : (bf16)0.f;
}

// --- head: out[b] = fc2f[b,:] . wout + bout ---------------------------------
__global__ void k_head(float* __restrict__ out, const float* __restrict__ fc2f,
                       const float* __restrict__ wout, const float* __restrict__ bout)
{
    int b = threadIdx.x;     // 1 x 256
    float acc = bout[0];
    for (int i = 0; i < 256; ++i) acc += fc2f[(size_t)b * 256 + i] * wout[i];
    out[b] = acc;
}

// ---------------------------------------------------------------------------
static void gemm(hipStream_t s, const bf16* A, const bf16* Bt, int M, int Kp,
                 int Np, int Nvalid, const float* bias, int relu,
                 const float* g, const float* bb, const float* mm, const float* vv,
                 float* outF, int ldF, bf16* outB, int ldB)
{
    int waves = (M >> 4) * (Np >> 6);    // one wave per 16x64 strip
    int blocks = (waves + 7) / 8;
    k_gemm<<<blocks, 256, 0, s>>>(A, Bt, M, Kp, Np, Nvalid, bias, relu,
                                  g, bb, mm, vv, outF, ldF, outB, ldB);
}

extern "C" void kernel_launch(void* const* d_in, const int* in_sizes, int n_in,
                              void* d_out, int out_size, void* d_ws, size_t ws_size,
                              hipStream_t stream)
{
    (void)in_sizes; (void)n_in; (void)out_size; (void)ws_size;
    auto F = [&](int i) { return (const float*)d_in[i]; };
    const float* x      = F(0);
    const int*   edge   = (const int*)d_in[1];
    const int*   target = (const int*)d_in[3];
    // params (setup_inputs insertion order)
    const int gl1w[5] = {4, 8, 12, 16, 20}, gl1b[5] = {5, 9, 13, 17, 21};
    const int gl2w[5] = {6, 10, 14, 18, 22}, gl2b[5] = {7, 11, 15, 19, 23};
    const int bnbase[5] = {24, 28, 32, 36, 40};
    const float* emb = F(44);
    const float* w1 = F(45); const float* b1 = F(46);
    const float* w2 = F(47); const float* b2 = F(48);
    const float* wxd = F(49); const float* bxd = F(50);
    const float* wxt = F(51); const float* bxt = F(52);
    const float* wf1 = F(53); const float* bf1 = F(54);
    const float* wf2 = F(55); const float* bf2 = F(56);
    const float* wo  = F(57); const float* bo  = F(58);

    // ---- workspace carve-out ----
    char* p = (char*)d_ws;
    auto take = [&](size_t bytes) -> void* {
        void* r = (void*)p; p += (bytes + 255) & ~(size_t)255; return r;
    };
    float* agg  = (float*)take(sizeof(float) * kN * kDimP);
    bf16* Abf   = (bf16*)take(sizeof(bf16) * kN * kDimP);
    bf16* t1bf  = (bf16*)take(sizeof(bf16) * kN * kDimP);
    float* hb0  = (float*)take(sizeof(float) * kN * kDimP);
    float* hb1  = (float*)take(sizeof(float) * kN * kDimP);
    float* hb2  = (float*)take(sizeof(float) * kN * kDimP);
    bf16* wl1[5]; bf16* wl2[5];
    for (int i = 0; i < 5; ++i) wl1[i] = (bf16*)take(sizeof(bf16) * 128 * 128);
    for (int i = 0; i < 5; ++i) wl2[i] = (bf16*)take(sizeof(bf16) * 128 * 128);
    bf16* wxd_t = (bf16*)take(sizeof(bf16) * 128 * 128);
    bf16* wxt_t = (bf16*)take(sizeof(bf16) * 128 * (kNfilt * kL2));
    bf16* wf1_t = (bf16*)take(sizeof(bf16) * 1024 * 256);
    bf16* wf2_t = (bf16*)take(sizeof(bf16) * 256 * 1024);
    bf16* Abig  = (bf16*)take(sizeof(bf16) * (kB * kNfilt) * kKc1);
    bf16* TabT  = (bf16*)take(sizeof(bf16) * 128 * kKc1);
    float* c1f  = (float*)take(sizeof(float) * kB * kNfilt * kDim);
    float* o1f  = (float*)take(sizeof(float) * kB * kNfilt * kDim);
    bf16* o2bf  = (bf16*)take(sizeof(bf16) * kB * (kNfilt * kL2));
    bf16* xg_bf = (bf16*)take(sizeof(bf16) * kB * kDimP);
    bf16* xc_bf = (bf16*)take(sizeof(bf16) * kB * 256);
    bf16* fc1bf = (bf16*)take(sizeof(bf16) * kB * 1024);
    float* fc2f = (float*)take(sizeof(float) * kB * 256);

    // ---- weight conversions (transposed + padded bf16) ----
    auto convW = [&](bf16* dst, const float* W, int K, int N, int Kp, int Np) {
        int tot = Np * Kp;
        k_convW<<<(tot + 255) / 256, 256, 0, stream>>>(dst, W, K, N, Kp, Np);
    };
    convW(wl1[0], F(gl1w[0]), kNFXD, kDim, 96, 128);
    for (int i = 1; i < 5; ++i) convW(wl1[i], F(gl1w[i]), kDim, kDim, 128, 128);
    for (int i = 0; i < 5; ++i) convW(wl2[i], F(gl2w[i]), kDim, kDim, 128, 128);
    convW(wxd_t, wxd, kDim, 128, 128, 128);
    convW(wxt_t, wxt, kNfilt * kL2, 128, kNfilt * kL2, 128);   // 3648 is a multiple of 32
    convW(wf1_t, wf1, 256, 1024, 256, 1024);
    convW(wf2_t, wf2, 1024, 256, 1024, 256);

    // ---- protein conv1 via vocab-26 factorization (WMMA GEMM) ----
    k_build_TabT<<<128, kKc1, 0, stream>>>(TabT, emb);
    k_build_A1<<<kB, 256, 0, stream>>>(Abig, w1, b1, target);
    gemm(stream, Abig, TabT, kB * kNfilt, kKc1, 128, kDim,
         nullptr, 0, nullptr, nullptr, nullptr, nullptr,
         c1f, kDim, nullptr, 0);

    // ---- 5 GIN layers: copy+scatter (agg), convert, two WMMA GEMMs ----
    float* hout[5] = {hb0, hb1, hb2, hb0, hb1};   // h3=hb2, h5=hb1 persist
    const float* hprev = x;
    int sstride = kNFXD, Ci = kNFXD, Kp = 96;
    for (int L = 0; L < 5; ++L) {
        k_copy<<<kN, 128, 0, stream>>>(agg, hprev, sstride, Ci);
        k_scatter<<<kE, 128, 0, stream>>>(agg, hprev, sstride, edge, Ci);
        int tot = kN * Kp;
        k_convA<<<(tot + 255) / 256, 256, 0, stream>>>(Abf, agg, kDimP, kN, Ci, Kp);
        gemm(stream, Abf, wl1[L], kN, Kp, 128, kDim,
             F(gl1b[L]), 1, nullptr, nullptr, nullptr, nullptr,
             nullptr, 0, t1bf, kDimP);
        gemm(stream, t1bf, wl2[L], kN, 128, 128, kDim,
             F(gl2b[L]), 1, F(bnbase[L]), F(bnbase[L] + 1), F(bnbase[L] + 2), F(bnbase[L] + 3),
             hout[L], kDimP, nullptr, 0);
        hprev = hout[L]; sstride = kDimP; Ci = kDim; Kp = 128;
        if (L == 2)  // attention 1 needs h3 + c1
            k_attn1<<<kB, 256, 0, stream>>>(o1f, c1f, hb2);
    }

    // ---- conv2 + attention 2 (fused, VALU) ----
    k_conv2_attn2<<<kB, 256, 0, stream>>>(o2bf, o1f, hb1, w2, b2);

    // ---- pooling + MLP head (WMMA GEMMs) ----
    k_pool<<<kB, 128, 0, stream>>>(xg_bf, hb1);
    gemm(stream, xg_bf, wxd_t, kB, 128, 128, 128,
         bxd, 1, nullptr, nullptr, nullptr, nullptr,
         nullptr, 0, xc_bf, 256);                       // cols 0..127 (relu)
    gemm(stream, o2bf, wxt_t, kB, kNfilt * kL2, 128, 128,
         bxt, 0, nullptr, nullptr, nullptr, nullptr,
         nullptr, 0, xc_bf + 128, 256);                 // cols 128..255 (no relu)
    gemm(stream, xc_bf, wf1_t, kB, 256, 1024, 1024,
         bf1, 1, nullptr, nullptr, nullptr, nullptr,
         nullptr, 0, fc1bf, 1024);
    gemm(stream, fc1bf, wf2_t, kB, 1024, 256, 256,
         bf2, 1, nullptr, nullptr, nullptr, nullptr,
         fc2f, 256, nullptr, 0);
    k_head<<<1, 256, 0, stream>>>((float*)d_out, fc2f, wo, bo);
}
